// GATGAE_17798344474651
// MI455X (gfx1250) — matrix operand order repaired and checked by reading, not compiled, not run on hardware
//
#include <hip/hip_runtime.h>
#include <hip/hip_bf16.h>
#include <math.h>

typedef __attribute__((ext_vector_type(16))) _Float16 v16h;
typedef __attribute__((ext_vector_type(8)))  float    v8f;

#define NNODES   50000
#define NEDGES   800000
#define ETOT     (NEDGES + NNODES)   // self-loops appended
#define IN_DIM   128
#define HIDC     64
#define HEADS    4
#define OUT_DIM  64
#define NEG_SLOPE 0.2f
#define EPS_F    1e-16f

// ---------------------------------------------------------------- utilities
__global__ void fill_f32(float* __restrict__ p, float v, int n) {
  int i = blockIdx.x * blockDim.x + threadIdx.x;
  if (i < n) p[i] = v;
}

__device__ __forceinline__ void edge_nodes(const int* __restrict__ ei, int e,
                                           int& s, int& d) {
  if (e < NEDGES) { s = ei[e]; d = ei[NEDGES + e]; }
  else            { s = d = e - NEDGES; }          // self loop
}

__device__ __forceinline__ void atomicMaxF32(float* addr, float v) {
  if (v >= 0.f) atomicMax((int*)addr, __float_as_int(v));
  else          atomicMin((unsigned int*)addr, __float_as_uint(v));
}

// ------------------------------------------------------------- frag packing
// A fragment (16-bit 16x32 ISA layout): lane holds row = 16*tm + (lane&15);
//   h[i], i<8  -> K = kt*32 + half*8 + i
//   h[i], i>=8 -> K = kt*32 + 16 + half*8 + (i-8)
// Flat packed order: ((tm*KT + kt)*32 + lane)*16 halfs -> 32B/lane, coalesced.
__global__ void pack_A(const float* __restrict__ A, _Float16* __restrict__ Ap,
                       int MT, int KT, int K) {
  int idx = blockIdx.x * blockDim.x + threadIdx.x;
  if (idx >= MT * KT * 32) return;
  int lane = idx & 31;
  int kt   = (idx >> 5) % KT;
  int tm   = (idx >> 5) / KT;
  int half = lane >> 4;
  const float* Arow = A + (size_t)(tm * 16 + (lane & 15)) * K + kt * 32;
  v16h v;
#pragma unroll
  for (int j = 0; j < 4; ++j) {
    float2 lo = *(const float2*)(Arow + half * 8 + 2 * j);
    float2 hi = *(const float2*)(Arow + 16 + half * 8 + 2 * j);
    v[2 * j]     = (_Float16)lo.x;  v[2 * j + 1] = (_Float16)lo.y;
    v[8 + 2 * j] = (_Float16)hi.x;  v[9 + 2 * j] = (_Float16)hi.y;
  }
  *(v16h*)(Ap + (size_t)idx * 16) = v;
}

// B fragment (16-bit 32x16 ISA layout): lane holds col = 16*ct + (lane&15);
//   h[i], i<8  -> K = kt*32 + half*8 + i
//   h[i], i>=8 -> K = kt*32 + 16 + half*8 + (i-8)
__global__ void pack_B(const float* __restrict__ B, _Float16* __restrict__ Bp,
                       int KT, int NT, int N) {
  int idx = blockIdx.x * blockDim.x + threadIdx.x;
  if (idx >= KT * NT * 32) return;
  int lane = idx & 31;
  int ct   = (idx >> 5) % NT;
  int kt   = (idx >> 5) / NT;
  int half = lane >> 4, c = lane & 15;
  v16h v;
#pragma unroll
  for (int i = 0; i < 16; ++i)
    v[i] = (_Float16)B[(size_t)(kt * 32 + (i < 8 ? half * 8 + i : 16 + half * 8 + (i - 8))) * N + ct * 16 + c];
  *(v16h*)(Bp + (size_t)idx * 16) = v;
}

// ---------------------------------------------------------------- WMMA GEMM
// C[M, TN*16] = A * B (+bias)(+ELU). Packed-f16 fragments, f32 accumulate.
// 256-thread block = 8 waves; packed B staged once in LDS; each wave owns a
// full 16 x (TN*16) output strip. Fragment loads are software-pipelined one
// step ahead so ds_load/global_load latency hides under the previous v_wmma.
template<int TN, bool ELU>
__global__ __launch_bounds__(256)
void wmma_gemm_packed(const _Float16* __restrict__ Ap,
                      const _Float16* __restrict__ Bp,
                      const float* __restrict__ bias, float* __restrict__ C,
                      int MT, int KT) {
  constexpr int N = TN * 16;
  extern __shared__ __align__(16) _Float16 sB[];   // KT*TN*512 halfs

  // cooperative global->LDS fill of the whole packed B (16B chunks)
  const int tid = threadIdx.x;
  const int nv = (KT * TN * 512) / 8;              // uint4 chunks
  const uint4* bsrc = (const uint4*)Bp;
  uint4* bdst = (uint4*)sB;
  for (int i = tid; i < nv; i += 256) bdst[i] = bsrc[i];
  __syncthreads();

  const int lane = tid & 31;
  const int bm = blockIdx.x * 8 + (tid >> 5);      // wave-uniform row tile
  if (bm < MT) {
    const _Float16* aPtr = Ap + ((size_t)bm * KT) * 512 + lane * 16;
    const _Float16* bPtr = sB + lane * 16;

    v8f acc[TN] = {};
    v16h a = *(const v16h*)aPtr;                   // prologue: first A frag
    for (int kt = 0; kt < KT; ++kt) {
      v16h aNext = a;
      if (kt + 1 < KT) aNext = *(const v16h*)(aPtr + (kt + 1) * 512);
      v16h b = *(const v16h*)(bPtr + (kt * TN) * 512);
#pragma unroll
      for (int t = 0; t < TN; ++t) {
        v16h bNext = b;
        if (t + 1 < TN) bNext = *(const v16h*)(bPtr + (kt * TN + t + 1) * 512);
        acc[t] = __builtin_amdgcn_wmma_f32_16x16x32_f16(
            false, a, false, b, (short)0, acc[t], false, false);
        b = bNext;
      }
      a = aNext;
    }

    const int half = lane >> 4, r16 = lane & 15;
#pragma unroll
    for (int t = 0; t < TN; ++t) {
      const int col = t * 16 + r16;
      const float bv = bias ? bias[col] : 0.f;
#pragma unroll
      for (int r = 0; r < 8; ++r) {
        const int rowOut = bm * 16 + r + 8 * half;  // C/D VGPR layout
        float v = acc[t][r] + bv;
        if (ELU) v = (v > 0.f) ? v : (__expf(v) - 1.f);
        C[(size_t)rowOut * N + col] = v;
      }
    }
  }
}

// ------------------------------------------------------- attention pipeline
__global__ void att_scores(const float* __restrict__ h,
                           const float* __restrict__ att_s,
                           const float* __restrict__ att_d,
                           float* __restrict__ a_s, float* __restrict__ a_d,
                           int n, int heads, int ch) {
  int idx = blockIdx.x * blockDim.x + threadIdx.x;
  if (idx >= n * heads) return;
  int node = idx / heads, hd = idx % heads;
  const float* hp = h + (size_t)node * heads * ch + hd * ch;
  const float* as = att_s + hd * ch;
  const float* ad = att_d + hd * ch;
  float s = 0.f, d = 0.f;
  for (int c = 0; c < ch; ++c) { float v = hp[c]; s += v * as[c]; d += v * ad[c]; }
  a_s[idx] = s; a_d[idx] = d;
}

__global__ void edge_max(const int* __restrict__ ei,
                         const float* __restrict__ a_s, const float* __restrict__ a_d,
                         float* __restrict__ amax, int heads) {
  int idx = blockIdx.x * blockDim.x + threadIdx.x;
  if (idx >= ETOT * heads) return;
  int e = idx / heads, hd = idx % heads;
  int s, d; edge_nodes(ei, e, s, d);
  float al = a_s[s * heads + hd] + a_d[d * heads + hd];
  al = (al > 0.f) ? al : NEG_SLOPE * al;
  atomicMaxF32(&amax[d * heads + hd], al);
}

__global__ void edge_exp(const int* __restrict__ ei,
                         const float* __restrict__ a_s, const float* __restrict__ a_d,
                         const float* __restrict__ amax,
                         float* __restrict__ alpha, float* __restrict__ denom,
                         int heads) {
  int idx = blockIdx.x * blockDim.x + threadIdx.x;
  if (idx >= ETOT * heads) return;
  int e = idx / heads, hd = idx % heads;
  int s, d; edge_nodes(ei, e, s, d);
  float al = a_s[s * heads + hd] + a_d[d * heads + hd];
  al = (al > 0.f) ? al : NEG_SLOPE * al;
  float w = __expf(al - amax[d * heads + hd]);
  alpha[idx] = w;
  atomicAdd(&denom[d * heads + hd], w);
}

// one thread per (edge, channel): coalesced gather + global_atomic_add_f32 scatter
__global__ void edge_scatter(const int* __restrict__ ei,
                             const float* __restrict__ h,
                             const float* __restrict__ alpha,
                             const float* __restrict__ denom,
                             float* __restrict__ out, int heads, int ch) {
  int idx = blockIdx.x * blockDim.x + threadIdx.x;
  const int HC = heads * ch;
  if (idx >= ETOT * HC) return;
  int e = idx / HC, c = idx % HC, hd = c / ch;
  int s, d; edge_nodes(ei, e, s, d);
  float coef = alpha[e * heads + hd] / (denom[d * heads + hd] + EPS_F);
  atomicAdd(&out[(size_t)d * HC + c], h[(size_t)s * HC + c] * coef);
}

__global__ void bias_elu(float* __restrict__ x, const float* __restrict__ b,
                         int n, int dim) {
  int i = blockIdx.x * blockDim.x + threadIdx.x;
  if (i >= n * dim) return;
  float v = x[i] + b[i % dim];
  x[i] = (v > 0.f) ? v : (__expf(v) - 1.f);
}

__global__ void bias_norm(const float* __restrict__ x, const float* __restrict__ b,
                          float* __restrict__ z, int n) {
  int node = blockIdx.x * blockDim.x + threadIdx.x;
  if (node >= n) return;
  float buf[OUT_DIM];
  float ss = 0.f;
  const float* xp = x + (size_t)node * OUT_DIM;
  for (int c = 0; c < OUT_DIM; ++c) { float v = xp[c] + b[c]; buf[c] = v; ss += v * v; }
  float scale = 1.f / fmaxf(sqrtf(ss), 1e-12f);
  float* zp = z + (size_t)node * OUT_DIM;
  for (int c = 0; c < OUT_DIM; ++c) zp[c] = buf[c] * scale;
}

// ---------------------------------------------------------------- launcher
extern "C" void kernel_launch(void* const* d_in, const int* in_sizes, int n_in,
                              void* d_out, int out_size, void* d_ws, size_t ws_size,
                              hipStream_t stream) {
  const float* x        = (const float*)d_in[0];
  const int*   ei       = (const int*)  d_in[1];
  const float* W1       = (const float*)d_in[2];
  const float* att_src1 = (const float*)d_in[3];
  const float* att_dst1 = (const float*)d_in[4];
  const float* b1       = (const float*)d_in[5];
  const float* W2       = (const float*)d_in[6];
  const float* att_src2 = (const float*)d_in[7];
  const float* att_dst2 = (const float*)d_in[8];
  const float* b2       = (const float*)d_in[9];
  const float* dW1      = (const float*)d_in[10];
  const float* db1      = (const float*)d_in[11];
  const float* dW2      = (const float*)d_in[12];
  const float* db2      = (const float*)d_in[13];

  float* zout = (float*)d_out;                       // [N, 64]
  float* xhat = zout + (size_t)NNODES * OUT_DIM;     // [N, 128]

  // workspace carve-up
  float* ws    = (float*)d_ws;
  float* h1    = ws;                                  // [N,256]
  float* out1  = h1    + (size_t)NNODES * 256;        // [N,256]
  float* alpha = out1  + (size_t)NNODES * 256;        // [ETOT,HEADS]
  float* a_s   = alpha + (size_t)ETOT * HEADS;
  float* a_d   = a_s   + (size_t)NNODES * HEADS;
  float* amax  = a_d   + (size_t)NNODES * HEADS;
  float* denom = amax  + (size_t)NNODES * HEADS;
  _Float16* Ap = (_Float16*)(denom + (size_t)NNODES * HEADS);  // [N*256] halfs max
  _Float16* Bp = Ap + (size_t)NNODES * 256;                    // [32768] halfs max
  // layer-2 / decoder temporaries reuse the h1 region
  float* h2   = h1;                                   // [N,64]
  float* out2 = h1 + (size_t)NNODES * 64;             // [N,64]
  float* tdec = h1 + (size_t)NNODES * 128;            // [N,64]

  const int MT = NNODES / 16;            // 3125 row tiles (exact)
  const int GB = (MT + 7) / 8;           // 8 waves / block
  auto cdiv = [](int a, int b) { return (a + b - 1) / b; };

  // ---- layer 1: h1 = x @ W1   [N,128]x[128,256]
  pack_A<<<cdiv(MT * 4 * 32, 256), 256, 0, stream>>>(x, Ap, MT, 4, IN_DIM);
  pack_B<<<cdiv(4 * 16 * 32, 256), 256, 0, stream>>>(W1, Bp, 4, 16, 256);
  wmma_gemm_packed<16, false><<<GB, 256, 4 * 16 * 512 * 2, stream>>>(
      Ap, Bp, nullptr, h1, MT, 4);
  att_scores<<<cdiv(NNODES * HEADS, 256), 256, 0, stream>>>(
      h1, att_src1, att_dst1, a_s, a_d, NNODES, HEADS, HIDC);
  fill_f32<<<cdiv(NNODES * HEADS, 256), 256, 0, stream>>>(amax, -INFINITY, NNODES * HEADS);
  fill_f32<<<cdiv(NNODES * HEADS, 256), 256, 0, stream>>>(denom, 0.f, NNODES * HEADS);
  fill_f32<<<cdiv(NNODES * 256, 256), 256, 0, stream>>>(out1, 0.f, NNODES * 256);
  edge_max<<<cdiv(ETOT * HEADS, 256), 256, 0, stream>>>(ei, a_s, a_d, amax, HEADS);
  edge_exp<<<cdiv(ETOT * HEADS, 256), 256, 0, stream>>>(ei, a_s, a_d, amax, alpha, denom, HEADS);
  edge_scatter<<<cdiv(ETOT * HEADS * HIDC, 256), 256, 0, stream>>>(
      ei, h1, alpha, denom, out1, HEADS, HIDC);
  bias_elu<<<cdiv(NNODES * 256, 256), 256, 0, stream>>>(out1, b1, NNODES, 256);

  // ---- layer 2: h2 = out1 @ W2   [N,256]x[256,64]
  pack_A<<<cdiv(MT * 8 * 32, 256), 256, 0, stream>>>(out1, Ap, MT, 8, 256);
  pack_B<<<cdiv(8 * 4 * 32, 256), 256, 0, stream>>>(W2, Bp, 8, 4, OUT_DIM);
  wmma_gemm_packed<4, false><<<GB, 256, 8 * 4 * 512 * 2, stream>>>(
      Ap, Bp, nullptr, h2, MT, 8);
  att_scores<<<cdiv(NNODES, 256), 256, 0, stream>>>(
      h2, att_src2, att_dst2, a_s, a_d, NNODES, 1, OUT_DIM);
  fill_f32<<<cdiv(NNODES, 256), 256, 0, stream>>>(amax, -INFINITY, NNODES);
  fill_f32<<<cdiv(NNODES, 256), 256, 0, stream>>>(denom, 0.f, NNODES);
  fill_f32<<<cdiv(NNODES * OUT_DIM, 256), 256, 0, stream>>>(out2, 0.f, NNODES * OUT_DIM);
  edge_max<<<cdiv(ETOT, 256), 256, 0, stream>>>(ei, a_s, a_d, amax, 1);
  edge_exp<<<cdiv(ETOT, 256), 256, 0, stream>>>(ei, a_s, a_d, amax, alpha, denom, 1);
  edge_scatter<<<cdiv(ETOT * OUT_DIM, 256), 256, 0, stream>>>(
      ei, h2, alpha, denom, out2, 1, OUT_DIM);

  // ---- z = normalize(out2 + b2)  ->  d_out[0 : N*64)
  bias_norm<<<cdiv(NNODES, 256), 256, 0, stream>>>(out2, b2, zout, NNODES);

  // ---- decoder: x_hat = ELU(z @ dW1 + db1) @ dW2 + db2
  pack_A<<<cdiv(MT * 2 * 32, 256), 256, 0, stream>>>(zout, Ap, MT, 2, OUT_DIM);
  pack_B<<<cdiv(2 * 4 * 32, 256), 256, 0, stream>>>(dW1, Bp, 2, 4, HIDC);
  wmma_gemm_packed<4, true><<<GB, 256, 2 * 4 * 512 * 2, stream>>>(
      Ap, Bp, db1, tdec, MT, 2);
  pack_A<<<cdiv(MT * 2 * 32, 256), 256, 0, stream>>>(tdec, Ap, MT, 2, HIDC);
  pack_B<<<cdiv(2 * 8 * 32, 256), 256, 0, stream>>>(dW2, Bp, 2, 8, IN_DIM);
  wmma_gemm_packed<8, false><<<GB, 256, 2 * 8 * 512 * 2, stream>>>(
      Ap, Bp, db2, xhat, MT, 2);
}
